// GNNPerformancePredictor_58325655880052
// MI455X (gfx1250) — matrix operand order repaired and checked by reading, not compiled, not run on hardware
//
#include <hip/hip_runtime.h>
#include <hip/hip_bf16.h>
#include <math.h>

// ---------------------------------------------------------------------------
// CDNA5 (gfx1250) wave32 WMMA types & helpers
// ---------------------------------------------------------------------------
typedef __attribute__((ext_vector_type(16))) _Float16 v16h;
typedef __attribute__((ext_vector_type(8)))  _Float16 v8h;
typedef __attribute__((ext_vector_type(8)))  float    v8f;
typedef __attribute__((ext_vector_type(4)))  unsigned int v4u;
typedef __attribute__((ext_vector_type(8)))  unsigned int v8u;

__device__ __forceinline__ v8f wmma16(v16h a, v16h b, v8f c) {
  // D = A(16x32 f16) * B(32x16 f16) + C(16x16 f32)
  return __builtin_amdgcn_wmma_f32_16x16x32_f16(
      /*neg_a=*/false, a, /*neg_b=*/false, b,
      /*c_mod=*/(short)0, c, /*reuse_a=*/false, /*reuse_b=*/false);
}

// A fragment (16x32, f16 source, row-major, ld = leading dim in elements).
// Per ISA 16-bit A layout: lanes 0-15 hold K {0..7,16..23}, lanes 16-31 hold
// K {8..15,24..31} for row M = lane&15.
__device__ __forceinline__ v16h load_frag_a_f16(const _Float16* p, int ld,
                                                int row0, int k0, int lane) {
  const _Float16* base = p + (size_t)(row0 + (lane & 15)) * ld + k0 + ((lane >> 4) << 3);
  v8h lo = *(const v8h*)(base);
  v8h hi = *(const v8h*)(base + 16);
  v16h r;
#pragma unroll
  for (int i = 0; i < 8; ++i) { r[i] = lo[i]; r[i + 8] = hi[i]; }
  return r;
}

// A fragment from f32 source (convert to f16 in registers).
__device__ __forceinline__ v16h load_frag_a_f32(const float* p, int ld,
                                                int row0, int k0, int lane) {
  const float* base = p + (size_t)(row0 + (lane & 15)) * ld + k0 + ((lane >> 4) << 3);
  v16h r;
#pragma unroll
  for (int i = 0; i < 8; ++i) {
    r[i]     = (_Float16)base[i];
    r[i + 8] = (_Float16)base[16 + i];
  }
  return r;
}

// B fragment (32x16). Weights are pre-transposed to Wt[N][K] (f16) so each
// lane (column n = n0 + lane&15) reads K {0..7,16..23} (lanes 0-15) or
// K {8..15,24..31} (lanes 16-31), matching the A fragment K order.
__device__ __forceinline__ v16h load_frag_b_f16(const _Float16* wt, int ldk,
                                                int n0, int k0, int lane) {
  const _Float16* base = wt + (size_t)(n0 + (lane & 15)) * ldk + k0 + ((lane >> 4) << 3);
  v8h lo = *(const v8h*)(base);
  v8h hi = *(const v8h*)(base + 16);
  v16h r;
#pragma unroll
  for (int i = 0; i < 8; ++i) { r[i] = lo[i]; r[i + 8] = hi[i]; }
  return r;
}

// C/D scatter: lane L holds col n0+(L&15), VGPR i holds row 8*(L>>4)+i.
__device__ __forceinline__ void store_frag_c(float* p, int ld, int row0, int n0,
                                             int lane, v8f c) {
  int col = n0 + (lane & 15);
  int r0  = row0 + ((lane >> 4) << 3);
#pragma unroll
  for (int i = 0; i < 8; ++i) p[(size_t)(r0 + i) * ld + col] = c[i];
}

// Float atomic-max via sign-aware integer punning (works on LDS & global).
__device__ __forceinline__ void atomicMaxF(float* addr, float v) {
  if (v >= 0.0f) atomicMax((int*)addr, __float_as_int(v));
  else           atomicMin((unsigned int*)addr, __float_as_uint(v));
}

__device__ __forceinline__ float relu(float v) { return v > 0.0f ? v : 0.0f; }
__device__ __forceinline__ float lrelu(float v) { return v > 0.0f ? v : 0.2f * v; }

// ---------------------------------------------------------------------------
// Model dimensions (fixed by the reference)
// ---------------------------------------------------------------------------
#define N_L   2048
#define N_H   64
#define E_L   32768
#define E_H   256
#define D_IN  256
#define NHEAD 4

// ---------------------------------------------------------------------------
// Weight convert+transpose: W[K][N] f32  ->  Wt[N][K] f16
// ---------------------------------------------------------------------------
__global__ void k_wt(const float* __restrict__ W, _Float16* __restrict__ Wt,
                     int K, int N) {
  int i = blockIdx.x * blockDim.x + threadIdx.x;
  if (i >= K * N) return;
  int k = i / N, n = i - k * N;
  Wt[(size_t)n * K + k] = (_Float16)W[i];
}

// ---------------------------------------------------------------------------
// Plain WMMA GEMM: C[M,N] = A[M,K](f32) @ Wt[N,K](f16).  block=256 (8 waves),
// grid = (M/16, N/128); each wave owns one 16x16 tile.
// ---------------------------------------------------------------------------
__global__ void k_gemm(const float* __restrict__ A, const _Float16* __restrict__ Wt,
                       float* __restrict__ C, int K, int N) {
  int lane = threadIdx.x & 31;
  int wave = threadIdx.x >> 5;
  int row0 = blockIdx.x * 16;
  int n0   = blockIdx.y * 128 + wave * 16;
  v8f acc = {};
  for (int k0 = 0; k0 < K; k0 += 32) {
    // prefetch next A K-tile into cache (global_prefetch_b8)
    if (k0 + 32 < K)
      __builtin_prefetch((const void*)(A + (size_t)(row0 + (lane & 15)) * K + k0 + 32), 0, 3);
    v16h a = load_frag_a_f32(A, K, row0, k0, lane);
    v16h b = load_frag_b_f16(Wt, K, n0, k0, lane);
    acc = wmma16(a, b, acc);
  }
  store_frag_c(C, N, row0, n0, lane, acc);
}

// ---------------------------------------------------------------------------
// Fused Linear(->128) + bias + LayerNorm + ReLU.  block=128 (4 waves),
// grid = M/16; the workgroup owns 16 full rows (N=128).
// ---------------------------------------------------------------------------
__global__ void k_gemm_ln_relu(const float* __restrict__ A, const _Float16* __restrict__ Wt,
                               const float* __restrict__ bias, const float* __restrict__ gamma,
                               const float* __restrict__ beta, float* __restrict__ out,
                               int K) {
  __shared__ float tile[16][128];
  __shared__ float ln_m[16], ln_r[16];
  int lane = threadIdx.x & 31;
  int wave = threadIdx.x >> 5;
  int row0 = blockIdx.x * 16;
  for (int t = wave; t < 8; t += 4) {       // 8 col tiles / 4 waves
    int n0 = t * 16;
    v8f acc = {};
    for (int k0 = 0; k0 < K; k0 += 32) {
      v16h a = load_frag_a_f32(A, K, row0, k0, lane);
      v16h b = load_frag_b_f16(Wt, K, n0, k0, lane);
      acc = wmma16(a, b, acc);
    }
    store_frag_c(&tile[0][0], 128, 0, n0, lane, acc);
  }
  __syncthreads();
  if (threadIdx.x < 16) {
    int r = threadIdx.x;
    float s = 0.f, s2 = 0.f;
    for (int c = 0; c < 128; ++c) {
      float v = tile[r][c] + bias[c];
      s += v; s2 += v * v;
    }
    float m = s * (1.f / 128.f);
    ln_m[r] = m;
    ln_r[r] = rsqrtf(s2 * (1.f / 128.f) - m * m + 1e-5f);
  }
  __syncthreads();
  for (int i = threadIdx.x; i < 16 * 128; i += 128) {
    int r = i >> 7, c = i & 127;
    float v = (tile[r][c] + bias[c] - ln_m[r]) * ln_r[r] * gamma[c] + beta[c];
    out[(size_t)(row0 + r) * 128 + c] = relu(v);
  }
}

// ---------------------------------------------------------------------------
// Fused per-layer head pipeline. One workgroup (256 threads, 8 waves) per
// layer node; entire 64-node head graph lives in LDS (~172 KB dynamic).
//   TDM tile load -> enc(GEMM+LN+ReLU) -> GAT(hg0)+ReLU -> GAT(hg1)+ReLU
//   -> node mean
// ---------------------------------------------------------------------------
#define HEAD_SMEM_BYTES (64*256*2 + 64*256*4 + 64*256*4 + 4*256*4 + 2*64*4 \
                         + 320*4*4 + 2*320*4)

__global__ void k_head(const float* __restrict__ head_x, const int* __restrict__ head_ei,
                       const _Float16* __restrict__ enc_wt, const float* __restrict__ enc_b,
                       const float* __restrict__ enc_g, const float* __restrict__ enc_be,
                       const _Float16* __restrict__ hg0_wt, const float* __restrict__ hg0_as,
                       const float* __restrict__ hg0_ad, const float* __restrict__ hg0_b,
                       const _Float16* __restrict__ hg1_wt, const float* __restrict__ hg1_as,
                       const float* __restrict__ hg1_ad, const float* __restrict__ hg1_b,
                       float* __restrict__ head_mean) {
  extern __shared__ char smem[];
  _Float16* xh  = (_Float16*)smem;                       // [64][256] f16 GEMM-A
  float* acc    = (float*)(smem + 64 * 256 * 2);         // [64][256] f32
  float* wx     = acc + 64 * 256;                        // [64][256] f32 (also TDM landing zone)
  float* al_s   = wx + 64 * 256;                         // [64*4]
  float* al_d   = al_s + 256;
  float* mb     = al_d + 256;
  float* db     = mb + 256;
  float* ln_m   = db + 256;                              // [64]
  float* ln_r   = ln_m + 64;                             // [64]
  float* alpha  = ln_r + 64;                             // [320*4]
  int*   esrc   = (int*)(alpha + 320 * 4);               // [320]
  int*   edst   = esrc + 320;                            // [320]

  const int tid  = threadIdx.x;
  const int lane = tid & 31;
  const int wave = tid >> 5;
  const int l    = blockIdx.x;
  const int ETOT = E_H + N_H;   // edges + self loops = 320

  // ---- TDM: DMA the 64x256 f32 tile of head_x into LDS (wx as staging) ------
  const float* xin = head_x + (size_t)l * N_H * D_IN;
  if (wave == 0) {
    unsigned long long gaddr = (unsigned long long)(uintptr_t)xin;
    unsigned int lds_off = (unsigned int)(uintptr_t)(void*)wx;  // low 32 bits = LDS byte offset
    // D# group 0 (128b): count=1 | lds_addr | global_addr[56:0] | type=2
    v4u g0;
    g0[0] = 1u;
    g0[1] = lds_off;
    g0[2] = (unsigned int)(gaddr & 0xFFFFFFFFu);
    g0[3] = ((unsigned int)((gaddr >> 32) & 0x01FFFFFFu)) | (2u << 30);
    // D# group 1 (256b): data_size=4B, tensor 256x64, tile 256x64, stride 256
    v8u g1;
    g1[0] = (2u << 16);        // workgroup_mask=0 | data_size=2 (4 bytes)
    g1[1] = (256u << 16);      // atomic_barrier_addr=0 | tensor_dim0[15:0]=256
    g1[2] = (64u << 16);       // tensor_dim0[31:16]=0 | tensor_dim1[15:0]=64
    g1[3] = (256u << 16);      // tensor_dim1[31:16]=0 | tile_dim0=256
    g1[4] = 64u;               // tile_dim1=64 | tile_dim2=0
    g1[5] = 256u;              // tensor_dim0_stride[31:0]=256
    g1[6] = 0u;                // stride0[47:32]=0 | tensor_dim1_stride[15:0]=0
    g1[7] = 0u;
    asm volatile("tensor_load_to_lds %0, %1" :: "s"(g0), "s"(g1) : "memory");
    __builtin_amdgcn_s_wait_tensorcnt(0);
  }

  // ---- edge list (concurrent with the TDM transfer) --------------------------
  const int* ei = head_ei + (size_t)l * 2 * E_H;
  for (int i = tid; i < ETOT; i += 256) {
    if (i < E_H) { esrc[i] = ei[i]; edst[i] = ei[E_H + i]; }
    else         { esrc[i] = i - E_H; edst[i] = i - E_H; }
  }
  __syncthreads();

  // ---- f32 -> f16 conversion into the GEMM-A buffer --------------------------
  for (int i = tid; i < N_H * D_IN; i += 256) xh[i] = (_Float16)wx[i];
  __syncthreads();

  // ---- encoder GEMM: acc[64][128] = xh[64][256] @ enc_wt ---------------------
  for (int t = wave; t < 32; t += 8) {
    int row0 = (t >> 3) << 4;
    int n0   = (t & 7) << 4;
    v8f c = {};
    for (int k0 = 0; k0 < 256; k0 += 32) {
      v16h a = load_frag_a_f16(xh, 256, row0, k0, lane);
      v16h b = load_frag_b_f16(enc_wt, 256, n0, k0, lane);
      c = wmma16(a, b, c);
    }
    store_frag_c(acc, 128, row0, n0, lane, c);
  }
  __syncthreads();

  // ---- bias + LayerNorm + ReLU -> xh[:, :128] --------------------------------
  if (tid < 64) {
    float s = 0.f, s2 = 0.f;
    for (int c = 0; c < 128; ++c) {
      float v = acc[tid * 128 + c] + enc_b[c];
      s += v; s2 += v * v;
    }
    float m = s * (1.f / 128.f);
    ln_m[tid] = m;
    ln_r[tid] = rsqrtf(s2 * (1.f / 128.f) - m * m + 1e-5f);
  }
  __syncthreads();
  for (int i = tid; i < 64 * 128; i += 256) {
    int r = i >> 7, c = i & 127;
    float v = (acc[i] + enc_b[c] - ln_m[r]) * ln_r[r] * enc_g[c] + enc_be[c];
    xh[r * 256 + c] = (_Float16)relu(v);
  }
  __syncthreads();

  // ---- two GAT layers (H=4, C=64, concat) ------------------------------------
  for (int gl = 0; gl < 2; ++gl) {
    const _Float16* wt = gl ? hg1_wt : hg0_wt;
    const float* asg   = gl ? hg1_as : hg0_as;
    const float* adg   = gl ? hg1_ad : hg0_ad;
    const float* bg    = gl ? hg1_b  : hg0_b;
    const int K        = gl ? 256 : 128;

    // wx[64][256] = xh[:, :K] @ wt
    for (int t = wave; t < 64; t += 8) {
      int row0 = (t >> 4) << 4;
      int n0   = (t & 15) << 4;
      v8f c = {};
      for (int k0 = 0; k0 < K; k0 += 32) {
        v16h a = load_frag_a_f16(xh, 256, row0, k0, lane);
        v16h b = load_frag_b_f16(wt, K, n0, k0, lane);
        c = wmma16(a, b, c);
      }
      store_frag_c(wx, 256, row0, n0, lane, c);
    }
    __syncthreads();

    // attention logits per (node, head); init max/denom; zero accumulator
    {
      int node = tid >> 2, h = tid & 3;
      float ss = 0.f, sd = 0.f;
      const float* w = wx + node * 256 + h * 64;
      for (int c = 0; c < 64; ++c) {
        ss += w[c] * asg[h * 64 + c];
        sd += w[c] * adg[h * 64 + c];
      }
      al_s[tid] = ss; al_d[tid] = sd;
      mb[tid] = -1e30f; db[tid] = 0.f;
    }
    for (int i = tid; i < 64 * 256; i += 256) acc[i] = 0.f;
    __syncthreads();

    // pass 1: leaky-relu logits + segment max (LDS atomics)
    for (int i = tid; i < ETOT * 4; i += 256) {
      int e = i >> 2, h = i & 3;
      float ev = lrelu(al_s[esrc[e] * 4 + h] + al_d[edst[e] * 4 + h]);
      alpha[i] = ev;
      atomicMaxF(&mb[edst[e] * 4 + h], ev);
    }
    __syncthreads();
    // pass 2: exp + segment sum
    for (int i = tid; i < ETOT * 4; i += 256) {
      int e = i >> 2, h = i & 3;
      float ex = expf(alpha[i] - mb[edst[e] * 4 + h]);
      alpha[i] = ex;
      atomicAdd(&db[edst[e] * 4 + h], ex);
    }
    __syncthreads();
    // pass 3: normalize
    for (int i = tid; i < ETOT * 4; i += 256) {
      int e = i >> 2, h = i & 3;
      alpha[i] = alpha[i] / (db[edst[e] * 4 + h] + 1e-16f);
    }
    __syncthreads();
    // pass 4: aggregate wx[src]*alpha into acc[dst]
    for (int j = tid; j < ETOT * 8; j += 256) {
      int e = j >> 3, chunk = j & 7;     // 8 chunks of 32 cols
      int c0 = chunk << 5;
      int h  = chunk >> 1;               // 64 cols per head
      float a = alpha[e * 4 + h];
      const float* ws = wx + esrc[e] * 256 + c0;
      float* od = acc + edst[e] * 256 + c0;
      for (int c = 0; c < 32; ++c) atomicAdd(&od[c], ws[c] * a);
    }
    __syncthreads();
    // bias + ReLU -> acc (f32) and xh (f16, next GEMM A)
    for (int i = tid; i < 64 * 256; i += 256) {
      float v = relu(acc[i] + bg[i & 255]);
      acc[i] = v;
      xh[i]  = (_Float16)v;
    }
    __syncthreads();
  }

  // ---- mean over the 64 head nodes -------------------------------------------
  {
    float s = 0.f;
    for (int r = 0; r < 64; ++r) s += acc[r * 256 + tid];
    head_mean[(size_t)l * 256 + tid] = s * (1.f / 64.f);
  }
}

// ---------------------------------------------------------------------------
// Layer-level GAT helpers (N=2048 nodes, E=32768 edges + self loops, C=128)
// ---------------------------------------------------------------------------
__global__ void k_al(const float* __restrict__ wx, const float* __restrict__ a_s,
                     const float* __restrict__ a_d, float* __restrict__ al_s,
                     float* __restrict__ al_d) {
  int i = blockIdx.x * blockDim.x + threadIdx.x;
  if (i >= N_L * NHEAD) return;
  int n = i >> 2, h = i & 3;
  const float* w = wx + (size_t)n * 512 + h * 128;
  float ss = 0.f, sd = 0.f;
  for (int c = 0; c < 128; ++c) {
    ss += w[c] * a_s[h * 128 + c];
    sd += w[c] * a_d[h * 128 + c];
  }
  al_s[i] = ss; al_d[i] = sd;
}

__global__ void k_fill(float* __restrict__ p, float v, int n) {
  int i = blockIdx.x * blockDim.x + threadIdx.x;
  if (i < n) p[i] = v;
}

__device__ __forceinline__ void layer_edge(const int* ei, int e, int& src, int& dst) {
  if (e < E_L) { src = ei[e]; dst = ei[E_L + e]; }
  else         { src = e - E_L; dst = e - E_L; }
}

__global__ void k_edge_max(const int* __restrict__ ei, const float* __restrict__ al_s,
                           const float* __restrict__ al_d, float* __restrict__ ev,
                           float* __restrict__ m) {
  int i = blockIdx.x * blockDim.x + threadIdx.x;
  if (i >= (E_L + N_L) * NHEAD) return;
  int e = i >> 2, h = i & 3, src, dst;
  layer_edge(ei, e, src, dst);
  float v = lrelu(al_s[src * 4 + h] + al_d[dst * 4 + h]);
  ev[i] = v;
  atomicMaxF(&m[dst * 4 + h], v);
}

__global__ void k_edge_exp(const int* __restrict__ ei, float* __restrict__ ev,
                           const float* __restrict__ m, float* __restrict__ denom) {
  int i = blockIdx.x * blockDim.x + threadIdx.x;
  if (i >= (E_L + N_L) * NHEAD) return;
  int e = i >> 2, h = i & 3, src, dst;
  layer_edge(ei, e, src, dst);
  float ex = expf(ev[i] - m[dst * 4 + h]);
  ev[i] = ex;
  atomicAdd(&denom[dst * 4 + h], ex);
}

__global__ void k_edge_agg(const int* __restrict__ ei, const float* __restrict__ ev,
                           const float* __restrict__ denom, const float* __restrict__ wx,
                           float* __restrict__ out) {
  int j = blockIdx.x * blockDim.x + threadIdx.x;
  if (j >= (E_L + N_L) * 16) return;
  int e = j >> 4, chunk = j & 15;       // 16 chunks of 32 cols (512 total)
  int c0 = chunk << 5;
  int h  = chunk >> 2;                  // 128 cols per head
  int src, dst;
  layer_edge(ei, e, src, dst);
  float a = ev[e * 4 + h] / (denom[dst * 4 + h] + 1e-16f);
  const float* ws = wx + (size_t)src * 512 + c0;
  float* od = out + (size_t)dst * 512 + c0;
  for (int c = 0; c < 32; ++c) atomicAdd(&od[c], ws[c] * a);
}

__global__ void k_bias_relu(float* __restrict__ x, const float* __restrict__ b, int n) {
  int i = blockIdx.x * blockDim.x + threadIdx.x;
  if (i < n) x[i] = relu(x[i] + b[i & 511]);
}

__global__ void k_concat(const float* __restrict__ lx, const float* __restrict__ hf,
                         float* __restrict__ cat) {
  int i = blockIdx.x * blockDim.x + threadIdx.x;
  if (i >= N_L * 256) return;
  int r = i >> 8, c = i & 255;
  cat[i] = (c < 128) ? lx[r * 128 + c] : hf[r * 128 + (c - 128)];
}

// ---------------------------------------------------------------------------
// Tail: global mean pool -> ga block -> o1 block -> o2 + sigmoid. One WG.
// ---------------------------------------------------------------------------
__global__ void k_tail(const float* __restrict__ x,
                       const float* __restrict__ ga_w, const float* __restrict__ ga_b,
                       const float* __restrict__ ga_g, const float* __restrict__ ga_be,
                       const float* __restrict__ o1_w, const float* __restrict__ o1_b,
                       const float* __restrict__ o1_g, const float* __restrict__ o1_be,
                       const float* __restrict__ o2_w, const float* __restrict__ o2_b,
                       float* __restrict__ out) {
  __shared__ float g[512], gv[256], hv[128], st[2];
  int t = threadIdx.x;
  {
    float s = 0.f;
    for (int r = 0; r < N_L; ++r) s += x[(size_t)r * 512 + t];
    g[t] = s * (1.f / (float)N_L);
  }
  __syncthreads();
  if (t < 256) {
    float s = ga_b[t];
    for (int k = 0; k < 512; ++k) s += g[k] * ga_w[k * 256 + t];
    gv[t] = s;
  }
  __syncthreads();
  if (t == 0) {
    float s = 0.f, s2 = 0.f;
    for (int k = 0; k < 256; ++k) { s += gv[k]; s2 += gv[k] * gv[k]; }
    float m = s * (1.f / 256.f);
    st[0] = m; st[1] = rsqrtf(s2 * (1.f / 256.f) - m * m + 1e-5f);
  }
  __syncthreads();
  if (t < 256) gv[t] = relu((gv[t] - st[0]) * st[1] * ga_g[t] + ga_be[t]);
  __syncthreads();
  if (t < 128) {
    float s = o1_b[t];
    for (int k = 0; k < 256; ++k) s += gv[k] * o1_w[k * 128 + t];
    hv[t] = s;
  }
  __syncthreads();
  if (t == 0) {
    float s = 0.f, s2 = 0.f;
    for (int k = 0; k < 128; ++k) { s += hv[k]; s2 += hv[k] * hv[k]; }
    float m = s * (1.f / 128.f);
    st[0] = m; st[1] = rsqrtf(s2 * (1.f / 128.f) - m * m + 1e-5f);
  }
  __syncthreads();
  if (t < 128) hv[t] = relu((hv[t] - st[0]) * st[1] * o1_g[t] + o1_be[t]);
  __syncthreads();
  if (t == 0) {
    float s = o2_b[0];
    for (int k = 0; k < 128; ++k) s += hv[k] * o2_w[k];
    out[0] = 1.f / (1.f + expf(-s));
  }
}

// ---------------------------------------------------------------------------
// Host orchestration
// ---------------------------------------------------------------------------
extern "C" void kernel_launch(void* const* d_in, const int* in_sizes, int n_in,
                              void* d_out, int out_size, void* d_ws, size_t ws_size,
                              hipStream_t stream) {
  (void)in_sizes; (void)n_in; (void)out_size; (void)ws_size;
  const float* layer_x  = (const float*)d_in[0];
  const float* head_x   = (const float*)d_in[1];
  const int*   layer_ei = (const int*)d_in[2];
  const int*   head_ei  = (const int*)d_in[3];
  const float* enc_w  = (const float*)d_in[4];
  const float* enc_b  = (const float*)d_in[5];
  const float* enc_g  = (const float*)d_in[6];
  const float* enc_be = (const float*)d_in[7];
  const float* hg0_w  = (const float*)d_in[8];
  const float* hg0_as = (const float*)d_in[9];
  const float* hg0_ad = (const float*)d_in[10];
  const float* hg0_b  = (const float*)d_in[11];
  const float* hg1_w  = (const float*)d_in[12];
  const float* hg1_as = (const float*)d_in[13];
  const float* hg1_ad = (const float*)d_in[14];
  const float* hg1_b  = (const float*)d_in[15];
  const float* ha_w   = (const float*)d_in[16];
  const float* ha_b   = (const float*)d_in[17];
  const float* ha_g   = (const float*)d_in[18];
  const float* ha_be  = (const float*)d_in[19];
  const float* le_w   = (const float*)d_in[20];
  const float* le_b   = (const float*)d_in[21];
  const float* le_g   = (const float*)d_in[22];
  const float* le_be  = (const float*)d_in[23];
  const float* lg0_w  = (const float*)d_in[24];
  const float* lg0_as = (const float*)d_in[25];
  const float* lg0_ad = (const float*)d_in[26];
  const float* lg0_b  = (const float*)d_in[27];
  const float* lg1_w  = (const float*)d_in[28];
  const float* lg1_as = (const float*)d_in[29];
  const float* lg1_ad = (const float*)d_in[30];
  const float* lg1_b  = (const float*)d_in[31];
  const float* lg2_w  = (const float*)d_in[32];
  const float* lg2_as = (const float*)d_in[33];
  const float* lg2_ad = (const float*)d_in[34];
  const float* lg2_b  = (const float*)d_in[35];
  const float* ga_w   = (const float*)d_in[36];
  const float* ga_b   = (const float*)d_in[37];
  const float* ga_g   = (const float*)d_in[38];
  const float* ga_be  = (const float*)d_in[39];
  const float* o1_w   = (const float*)d_in[40];
  const float* o1_b   = (const float*)d_in[41];
  const float* o1_g   = (const float*)d_in[42];
  const float* o1_be  = (const float*)d_in[43];
  const float* o2_w   = (const float*)d_in[44];
  const float* o2_b   = (const float*)d_in[45];

  // workspace carve (bump allocator, 256B aligned)
  char* base = (char*)d_ws;
  size_t off = 0;
  auto alloc = [&](size_t bytes) -> void* {
    void* p = base + off;
    off += (bytes + 255) & ~(size_t)255;
    return p;
  };
  _Float16* enc_wt = (_Float16*)alloc(256 * 128 * 2);
  _Float16* hg0_wt = (_Float16*)alloc(128 * 256 * 2);
  _Float16* hg1_wt = (_Float16*)alloc(256 * 256 * 2);
  _Float16* ha_wt  = (_Float16*)alloc(256 * 128 * 2);
  _Float16* le_wt  = (_Float16*)alloc(256 * 128 * 2);
  _Float16* lg0_wt = (_Float16*)alloc(128 * 512 * 2);
  _Float16* lg1_wt = (_Float16*)alloc(512 * 512 * 2);
  _Float16* lg2_wt = (_Float16*)alloc(512 * 512 * 2);
  float* lx        = (float*)alloc((size_t)N_L * 128 * 4);
  float* head_mean = (float*)alloc((size_t)N_L * 256 * 4);
  float* head_feat = (float*)alloc((size_t)N_L * 128 * 4);
  float* cat       = (float*)alloc((size_t)N_L * 256 * 4);
  float* x0        = (float*)alloc((size_t)N_L * 128 * 4);
  float* wxb       = (float*)alloc((size_t)N_L * 512 * 4);
  float* alsb      = (float*)alloc((size_t)N_L * 4 * 4);
  float* aldb      = (float*)alloc((size_t)N_L * 4 * 4);
  float* mbuf      = (float*)alloc((size_t)N_L * 4 * 4);
  float* dbuf      = (float*)alloc((size_t)N_L * 4 * 4);
  float* evb       = (float*)alloc((size_t)(E_L + N_L) * 4 * 4);
  float* x1        = (float*)alloc((size_t)N_L * 512 * 4);
  float* x2        = (float*)alloc((size_t)N_L * 512 * 4);
  float* x3        = (float*)alloc((size_t)N_L * 512 * 4);

  // 1. weights -> f16, transposed [N][K]
  auto cvt = [&](const float* W, _Float16* Wt, int K, int N) {
    k_wt<<<(K * N + 255) / 256, 256, 0, stream>>>(W, Wt, K, N);
  };
  cvt(enc_w, enc_wt, 256, 128);
  cvt(hg0_w, hg0_wt, 128, 256);
  cvt(hg1_w, hg1_wt, 256, 256);
  cvt(ha_w,  ha_wt,  256, 128);
  cvt(le_w,  le_wt,  256, 128);
  cvt(lg0_w, lg0_wt, 128, 512);
  cvt(lg1_w, lg1_wt, 512, 512);
  cvt(lg2_w, lg2_wt, 512, 512);

  // 2. encoder on layer_x
  k_gemm_ln_relu<<<N_L / 16, 128, 0, stream>>>(layer_x, enc_wt, enc_b, enc_g, enc_be, lx, 256);

  // 3. fused per-layer head pipeline (TDM + everything in LDS)
  k_head<<<N_L, 256, HEAD_SMEM_BYTES, stream>>>(
      head_x, head_ei, enc_wt, enc_b, enc_g, enc_be,
      hg0_wt, hg0_as, hg0_ad, hg0_b,
      hg1_wt, hg1_as, hg1_ad, hg1_b, head_mean);

  // 4. head aggregate block + concat + layer encoder block
  k_gemm_ln_relu<<<N_L / 16, 128, 0, stream>>>(head_mean, ha_wt, ha_b, ha_g, ha_be, head_feat, 256);
  k_concat<<<(N_L * 256 + 255) / 256, 256, 0, stream>>>(lx, head_feat, cat);
  k_gemm_ln_relu<<<N_L / 16, 128, 0, stream>>>(cat, le_wt, le_b, le_g, le_be, x0, 256);

  // 5. three layer-level GATs
  auto gat = [&](const float* xin, int K, const _Float16* wt, const float* as_,
                 const float* ad_, const float* bias, float* xout) {
    k_gemm<<<dim3(N_L / 16, 4), 256, 0, stream>>>(xin, wt, wxb, K, 512);
    k_al<<<(N_L * 4 + 255) / 256, 256, 0, stream>>>(wxb, as_, ad_, alsb, aldb);
    k_fill<<<(N_L * 4 + 255) / 256, 256, 0, stream>>>(mbuf, -1e30f, N_L * 4);
    k_fill<<<(N_L * 4 + 255) / 256, 256, 0, stream>>>(dbuf, 0.f, N_L * 4);
    k_fill<<<(N_L * 512 + 255) / 256, 256, 0, stream>>>(xout, 0.f, N_L * 512);
    int et4 = (E_L + N_L) * 4;
    k_edge_max<<<(et4 + 255) / 256, 256, 0, stream>>>(layer_ei, alsb, aldb, evb, mbuf);
    k_edge_exp<<<(et4 + 255) / 256, 256, 0, stream>>>(layer_ei, evb, mbuf, dbuf);
    int et16 = (E_L + N_L) * 16;
    k_edge_agg<<<(et16 + 255) / 256, 256, 0, stream>>>(layer_ei, evb, dbuf, wxb, xout);
    k_bias_relu<<<(N_L * 512 + 255) / 256, 256, 0, stream>>>(xout, bias, N_L * 512);
  };
  gat(x0, 128, lg0_wt, lg0_as, lg0_ad, lg0_b, x1);
  gat(x1, 512, lg1_wt, lg1_as, lg1_ad, lg1_b, x2);
  gat(x2, 512, lg2_wt, lg2_as, lg2_ad, lg2_b, x3);

  // 6. pool + ga/o1/o2 head
  k_tail<<<1, 512, 0, stream>>>(x3, ga_w, ga_b, ga_g, ga_be,
                                o1_w, o1_b, o1_g, o1_be, o2_w, o2_b, (float*)d_out);
}
